// DilatedAttention_91130616087174
// MI455X (gfx1250) — compile-verified, matrix operand order checked by reading, not run on hardware
//
#include <hip/hip_runtime.h>

typedef _Float16 half_t;
typedef __attribute__((ext_vector_type(16))) _Float16 v16h;
typedef __attribute__((ext_vector_type(8)))  _Float16 v8h;
typedef __attribute__((ext_vector_type(4)))  _Float16 v4h;
typedef __attribute__((ext_vector_type(8)))  float    v8f;
typedef __attribute__((ext_vector_type(4)))  float    v4f;
typedef __attribute__((ext_vector_type(4)))  int      v4i;

#define AS1 __attribute__((address_space(1)))
#define AS3 __attribute__((address_space(3)))

#if __has_builtin(__builtin_amdgcn_global_load_async_to_lds_b128)
#define HAVE_ASYNC_LDS 1
#else
#define HAVE_ASYNC_LDS 0
#endif

#define DIM      1024
#define SEG      512      // dilated rows (tokens) per segment
#define BM       32       // query rows per workgroup
#define BN       64       // keys per streamed block
#define NTHREADS 256      // 8 wave32
#define LDQ      1032     // halves  (1024 + 8 pad)
#define LDK      1032     // halves
#define LDVT     72       // halves  (64 + 8 pad), V^T rows are d-indices
#define LDS_S    520      // floats  (512 + 8 pad)
#define LDP      520      // halves
#define SCALE    0.03125f // 1/sqrt(1024)

// ---- LDS layout (bytes) ----
#define OFF_S    0
#define SZ_S     (BM * LDS_S * 4)        // 66560
#define OFF_Q    (OFF_S + SZ_S)
#define SZ_Q     (BM * LDQ * 2)          // 66048 (reused as f16 probs in pass 2)
#define OFF_KV   (OFF_Q + SZ_Q)
#define SZ_KV    (DIM * LDVT * 2)        // 147456 (>= 64*LDK*2 = 132096, >= 128KB Q stage)
#define OFF_RED  (OFF_KV + SZ_KV)
#define SZ_RED   (BM * 8 * 4)
#define OFF_INV  (OFF_RED + SZ_RED)
#define SZ_INV   (BM * 4)
#define LDS_TOTAL (OFF_INV + SZ_INV)     // 281216 B  (< 320 KB)

// flat->LDS(as3) is a low-32-bit truncation per the ISA aperture rules;
// flat->global(as1) is value-preserving.  Builtin params are typed int4
// pointers (C++ has no implicit void* conversion), so cast to v4i*.
__device__ __forceinline__ AS3 v4i* as_lds_v4i(void* p) {
  return (AS3 v4i*)(unsigned int)(unsigned long long)p;
}
__device__ __forceinline__ AS1 v4i* as_global_v4i(const void* p) {
  return (AS1 v4i*)(unsigned long long)p;
}

// A-fragment loader for V_WMMA_F32_16X16X32_F16, per ISA 16-bit A 16x32 layout:
// lanes 0-15:  VGPR0-3 = K 0..7,  VGPR4-7 = K 16..23   (row M = lane)
// lanes 16-31: VGPR0-3 = K 8..15, VGPR4-7 = K 24..31   (row M = lane-16)
// The B operand of D = A*B uses the same layout applied to B^T (row-major B^T source).
__device__ __forceinline__ v16h load_fragA(const half_t* __restrict__ base, int ld,
                                           int row0, int col0, int lane) {
  const int m  = lane & 15;
  const int hi = (lane >> 4) << 3;            // 0 or 8
  const half_t* p = base + (row0 + m) * ld + col0 + hi;
  v8h lo = *(const v8h*)(p);                  // K = hi .. hi+7
  v8h hv = *(const v8h*)(p + 16);             // K = 16+hi .. 16+hi+7
  v16h r;
#pragma unroll
  for (int i = 0; i < 8; ++i) { r[i] = lo[i]; r[i + 8] = hv[i]; }
  return r;
}

__device__ __forceinline__ v8f wmma_f16(v16h a, v16h b, v8f c) {
  return __builtin_amdgcn_wmma_f32_16x16x32_f16(false, a, false, b, (short)0, c,
                                                false, false);
}

__global__ __launch_bounds__(NTHREADS)
void DilatedAttention_91130616087174_kernel(const float* __restrict__ x,
                                            float* __restrict__ out) {
  extern __shared__ char smem[];
  float*  sS   = (float*)(smem + OFF_S);    // 32 x 512 fp32 scores
  half_t* sQ   = (half_t*)(smem + OFF_Q);   // 32 x 1024 f16 queries (pass 1)
  half_t* sP   = (half_t*)(smem + OFF_Q);   // 32 x 512 f16 probs   (pass 2, reuses Q)
  half_t* sK   = (half_t*)(smem + OFF_KV);  // 64 x 1024 f16 keys   (pass 1 staging)
  half_t* sVt  = (half_t*)(smem + OFF_KV);  // 1024 x 64 f16 V^T    (pass 2 staging)
  float*  sRed = (float*)(smem + OFF_RED);  // 32 x 8 reduction scratch
  float*  sInv = (float*)(smem + OFF_INV);  // 32 row 1/sum

  const int tid  = threadIdx.x;
  const int w    = tid >> 5;                // wave id (wave32)
  const int lane = tid & 31;
  const int bs   = blockIdx.x >> 4;         // b*8 + segment, 0..31
  const int mblk = blockIdx.x & 15;         // 16 M-blocks of 32 rows

  const size_t in_seg = (size_t)bs * 1024 * DIM;   // segment base row * DIM

  // ---------------- load Q tile (dilated rows), fp32 -> f16 ----------------
  {
    const int r  = tid >> 3;                 // 0..31
    const int d0 = (tid & 7) * 128;
    const float* src = x + in_seg + (size_t)(2 * (mblk * BM + r)) * DIM + d0;
    half_t* dst = sQ + r * LDQ + d0;
#if HAVE_ASYNC_LDS
    // CDNA5 async path: DMA fp32 Q into the (not-yet-used) K/V^T staging region,
    // then convert LDS->LDS to f16.  global_load_async_to_lds_b128 + ASYNCcnt.
    float* stage = (float*)(smem + OFF_KV) + r * 1024 + d0;
#pragma unroll
    for (int i = 0; i < 128; i += 4) {
      __builtin_amdgcn_global_load_async_to_lds_b128(
          as_global_v4i(src + i), as_lds_v4i(stage + i), 0, 0);
    }
#if __has_builtin(__builtin_amdgcn_s_wait_asynccnt)
    __builtin_amdgcn_s_wait_asynccnt(0);
#else
    asm volatile("s_wait_asynccnt 0x0" ::: "memory");
#endif
    __syncthreads();
#pragma unroll 8
    for (int i = 0; i < 128; i += 4) {
      v4f f = *(const v4f*)(stage + i);
      v4h h; h[0] = (half_t)f[0]; h[1] = (half_t)f[1];
             h[2] = (half_t)f[2]; h[3] = (half_t)f[3];
      *(v4h*)(dst + i) = h;
    }
#else
#pragma unroll 8
    for (int i = 0; i < 128; i += 4) {
      v4f f = *(const v4f*)(src + i);
      v4h h; h[0] = (half_t)f[0]; h[1] = (half_t)f[1];
             h[2] = (half_t)f[2]; h[3] = (half_t)f[3];
      *(v4h*)(dst + i) = h;
    }
#endif
  }

  // ---------------- pass 1: S = (Q K^T) * scale ----------------
  const int mi  = w & 1;                     // wave's M tile (0/1)
  const int nq  = w >> 1;                    // wave's key tile within block (0..3)
  const int m0  = mi * 16;
  const int kl0 = nq * 16;
  const int n   = lane & 15;
  const int mo  = (lane >> 4) * 8;           // C/D layout: lanes 16-31 hold M+8

  for (int kb = 0; kb < SEG / BN; ++kb) {
    __syncthreads();                         // protect sK overwrite (also fences Q stage)
    {
      const int r  = tid >> 2;               // 0..63 key row in block
      const int d0 = (tid & 3) * 256;
      const float* src = x + in_seg + (size_t)(2 * (kb * BN + r)) * DIM + d0;
      half_t* dst = sK + r * LDK + d0;
      if (kb + 1 < SEG / BN)                 // global_prefetch_b8 for next K block
        __builtin_prefetch(x + in_seg + (size_t)(2 * ((kb + 1) * BN + r)) * DIM + d0, 0, 1);
#pragma unroll 8
      for (int i = 0; i < 256; i += 4) {
        v4f f = *(const v4f*)(src + i);
        v4h h; h[0] = (half_t)f[0]; h[1] = (half_t)f[1];
               h[2] = (half_t)f[2]; h[3] = (half_t)f[3];
        *(v4h*)(dst + i) = h;
      }
    }
    __syncthreads();

    v8f acc = {};
#pragma unroll 4
    for (int c = 0; c < DIM / 32; ++c) {     // 32 WMMAs over the 1024-deep K dim
      v16h a = load_fragA(sQ, LDQ, m0, c * 32, lane);
      v16h b = load_fragA(sK, LDK, kl0, c * 32, lane);  // B^T = K tile (row-major)
      acc = wmma_f16(a, b, acc);
    }
    const int nc = kb * BN + nq * 16 + n;
#pragma unroll
    for (int v = 0; v < 8; ++v)
      sS[(m0 + mo + v) * LDS_S + nc] = acc[v] * SCALE;
  }
  __syncthreads();

  // ---------------- softmax over each row of 512 (8 threads / row) ----------------
  {
    const int r = tid >> 3, c = tid & 7;
    float* rowS = sS + r * LDS_S + c * 64;
    float m = -3.0e38f;
#pragma unroll 8
    for (int j = 0; j < 64; ++j) m = fmaxf(m, rowS[j]);
    sRed[r * 8 + c] = m;
    __syncthreads();
    float rm = sRed[r * 8];
#pragma unroll
    for (int i = 1; i < 8; ++i) rm = fmaxf(rm, sRed[r * 8 + i]);
    __syncthreads();                         // all reads of sRed done before reuse
    half_t* rowP = sP + r * LDP + c * 64;    // f16 probs overwrite dead Q region
    float sum = 0.f;
#pragma unroll 8
    for (int j = 0; j < 64; ++j) {
      float e = __expf(rowS[j] - rm);
      sum += e;
      rowP[j] = (half_t)e;                   // unnormalized; 1/sum folded at store
    }
    sRed[r * 8 + c] = sum;
    __syncthreads();
    if (c == 0) {
      float t = 0.f;
#pragma unroll
      for (int i = 0; i < 8; ++i) t += sRed[r * 8 + i];
      sInv[r] = 1.0f / t;
    }
  }

  // ---------------- pass 2: O = P V  (wave w owns output cols [128w, 128w+128)) ----------------
  v8f oacc[16] = {};                          // [mi*8 + ni] : 2 M-tiles x 8 N-tiles

  for (int kb = 0; kb < SEG / BN; ++kb) {
    __syncthreads();                          // protect sVt overwrite (also fences sInv/sP)
    {
      const int k  = tid >> 2;                // 0..63 local key
      const int d0 = (tid & 3) * 256;
      const float* src = x + in_seg + (size_t)(2 * (kb * BN + k)) * DIM + d0;
#pragma unroll 4
      for (int i = 0; i < 256; i += 4) {      // coalesced b128 read, transposed b16 scatter
        v4f f = *(const v4f*)(src + i);
        sVt[(d0 + i + 0) * LDVT + k] = (half_t)f[0];
        sVt[(d0 + i + 1) * LDVT + k] = (half_t)f[1];
        sVt[(d0 + i + 2) * LDVT + k] = (half_t)f[2];
        sVt[(d0 + i + 3) * LDVT + k] = (half_t)f[3];
      }
    }
    __syncthreads();

#pragma unroll
    for (int kk = 0; kk < 2; ++kk) {          // two 32-key chunks per block
      v16h a0 = load_fragA(sP, LDP, 0,  kb * BN + kk * 32, lane);
      v16h a1 = load_fragA(sP, LDP, 16, kb * BN + kk * 32, lane);
#pragma unroll
      for (int ni = 0; ni < 8; ++ni) {
        v16h b = load_fragA(sVt, LDVT, w * 128 + ni * 16, kk * 32, lane); // B^T = V^T
        oacc[ni]     = wmma_f16(a0, b, oacc[ni]);
        oacc[8 + ni] = wmma_f16(a1, b, oacc[8 + ni]);
      }
    }
  }

  // ---------------- writeout, normalized by 1/rowsum ----------------
  {
    const size_t orow0 = (size_t)bs * 512 + (size_t)mblk * BM;
#pragma unroll
    for (int mt = 0; mt < 2; ++mt)
#pragma unroll
      for (int ni = 0; ni < 8; ++ni) {
        v8f a = oacc[mt * 8 + ni];
#pragma unroll
        for (int v = 0; v < 8; ++v) {
          const int m = mt * 16 + mo + v;
          out[(orow0 + m) * DIM + w * 128 + ni * 16 + n] = a[v] * sInv[m];
        }
      }
  }
}

extern "C" void kernel_launch(void* const* d_in, const int* in_sizes, int n_in,
                              void* d_out, int out_size, void* d_ws, size_t ws_size,
                              hipStream_t stream) {
  (void)in_sizes; (void)n_in; (void)out_size; (void)d_ws; (void)ws_size;
  const float* x = (const float*)d_in[0];
  float* out = (float*)d_out;
  dim3 grid(32 * 16);        // 32 (batch,segment) pairs x 16 M-blocks
  dim3 block(NTHREADS);
  hipLaunchKernelGGL(DilatedAttention_91130616087174_kernel, grid, block,
                     LDS_TOTAL, stream, x, out);
}
